// RNN_38981123178621
// MI455X (gfx1250) — matrix operand order repaired and checked by reading, not compiled, not run on hardware
//
#include <hip/hip_runtime.h>
#include <hip/hip_bf16.h>
#include <stdint.h>

// Problem constants: B=64, T=512, D=256, U=256, R=4
#define NB 64
#define NT 512
#define ND 256
#define NU 256
#define BU (NB*NU)          // 16384
#define OUT_MAIN (NB*NT*NU) // 8388608
#define OUT_LO   (OUT_MAIN)
#define OUT_LS   (OUT_MAIN + BU)

// padded LDS row stride (u16 elems): 264*2 = 528 B = 132 dwords -> bank stride 4 (conflict-free)
// matches TDM padding: 512B row (128 dwords, pad_interval code 6) + 16B pad (4 dwords, code 3)
#define SP 264
#define PLANE (64*SP)

typedef __bf16 bf16x16 __attribute__((ext_vector_type(16)));
typedef float  f32x8   __attribute__((ext_vector_type(8)));
typedef unsigned short u16;
typedef u16 u16x8 __attribute__((ext_vector_type(8)));
typedef unsigned int v4u __attribute__((ext_vector_type(4)));
typedef int v4i __attribute__((ext_vector_type(4)));
typedef int v8i __attribute__((ext_vector_type(8)));

union BF16Vec {
    bf16x16 v;
    u16x8   h[2];
    u16     u[16];
};

__device__ __forceinline__ u16 f2bf(float f) {
    uint32_t x = __float_as_uint(f);
    uint32_t r = x + 0x7FFFu + ((x >> 16) & 1u);   // round-to-nearest-even
    return (u16)(r >> 16);
}
__device__ __forceinline__ float bf2f(u16 s) {
    return __uint_as_float(((uint32_t)s) << 16);
}
__device__ __forceinline__ float sigm(float x) { return 1.0f / (1.0f + __expf(-x)); }
__device__ __forceinline__ float tanh_(float x) { return 1.0f - 2.0f / (1.0f + __expf(2.0f * x)); }

// ---- Tensor Data Mover: 2D bf16 tile (256 cols x rows) global -> LDS with row padding ----
// D# per CDNA5 ISA ch.8: group0 {count, lds_addr, global_addr, type=2},
// group1 {data_size=2B, pad 4dw per 128dw, tensor/tile dims, dim0 stride}, groups2/3 = 0 (2D).
__device__ __forceinline__ void tdm_load_2d(uint32_t lds_off, const void* gaddr, uint32_t rows) {
    uint64_t ga = (uint64_t)(uintptr_t)gaddr;
    v4u g0;
    g0.x = 1u;                                                  // count=1, user mode
    g0.y = lds_off;                                             // LDS byte address
    g0.z = (uint32_t)ga;                                        // global_addr[31:0]
    g0.w = (uint32_t)((ga >> 32) & 0x1FFFFFFu) | 0x80000000u;   // addr[56:32] | type=2
    v8i g1;
    g1[0] = (int)((1u << 16) | (1u << 20) | (6u << 22) | (3u << 25)); // dsz=2B,pad_en,intv=128dw,amt=4dw
    g1[1] = (int)(256u << 16);           // tensor_dim0 = 256 (elements)
    g1[2] = (int)((rows & 0xFFFFu) << 16); // tensor_dim1 = rows
    g1[3] = (int)(256u << 16);           // tile_dim0 = 256
    g1[4] = (int)(rows & 0xFFFFu);       // tile_dim1 = rows, tile_dim2 = 0
    g1[5] = 256;                         // tensor_dim0_stride = 256 (elements)
    g1[6] = 0;
    g1[7] = 0;
    v4i gz = {0, 0, 0, 0};
#if defined(__clang_major__) && (__clang_major__ >= 23)
    v8i gz8 = {0, 0, 0, 0, 0, 0, 0, 0};
    __builtin_amdgcn_tensor_load_to_lds(g0, g1, gz, gz, gz8, 0);
#else
    __builtin_amdgcn_tensor_load_to_lds(g0, g1, gz, gz, 0);
#endif
}

// A operand: 16x32 bf16 tile, rows m0..m0+15, K = kk*32..kk*32+31; row-major src
__device__ __forceinline__ BF16Vec load_a16(const u16* base, int stride, int m0, int kk) {
    int lane = threadIdx.x & 31;
    int hl = lane >> 4;
    const u16* p = base + (m0 + (lane & 15)) * stride + kk * 32 + hl * 8;
    BF16Vec r;
    r.h[0] = *(const u16x8*)(p);
    r.h[1] = *(const u16x8*)(p + 16);
    return r;
}
// B operand: 32x16 bf16 tile from weights stored TRANSPOSED as [N][K]
__device__ __forceinline__ BF16Vec load_b16(const u16* bt, int stride, int n0, int kk) {
    int lane = threadIdx.x & 31;
    int hl = lane >> 4;
    const u16* p = bt + (n0 + (lane & 15)) * stride + kk * 32 + hl * 8;
    BF16Vec r;
    r.h[0] = *(const u16x8*)(p);
    r.h[1] = *(const u16x8*)(p + 16);
    return r;
}
// A operand = sum of 4 bf16 state planes (h_sum), summed in f32, repacked bf16
__device__ __forceinline__ BF16Vec load_a_sum4(const u16* Sb, int m0, int kk) {
    int lane = threadIdx.x & 31;
    int hl = lane >> 4;
    int row = m0 + (lane & 15);
    int c0 = kk * 32 + hl * 8;
    BF16Vec r;
#pragma unroll
    for (int j = 0; j < 8; ++j) {
        float fa = 0.f, fb = 0.f;
#pragma unroll
        for (int k = 0; k < 4; ++k) {
            const u16* pl = Sb + k * PLANE + row * SP;
            fa += bf2f(pl[c0 + j]);
            fb += bf2f(pl[c0 + 16 + j]);
        }
        r.u[j]     = f2bf(fa);
        r.u[8 + j] = f2bf(fb);
    }
    return r;
}
__device__ __forceinline__ f32x8 wmma_bf16(BF16Vec a, BF16Vec b, f32x8 c) {
    return __builtin_amdgcn_wmma_f32_16x16x32_bf16(false, a.v, false, b.v, (short)0, c, false, false);
}

// ---------------- setup kernels ----------------
__global__ void k_cvt_x(const float* __restrict__ x, u16* __restrict__ Xb) {
    int idx = blockIdx.x * 256 + threadIdx.x;        // < 32768*256
    int m = idx >> 8, k = idx & 255;
    int b = m & 63, t = m >> 6;
    Xb[idx] = f2bf(x[(b * NT + t) * ND + k]);
}
__global__ void k_cvt_w(const float* Wz, const float* Wr, const float* Wh,
                        const float* Uz, const float* Ur, const float* Uh,
                        const float* bz, const float* br, const float* bh,
                        u16* Wt, float* bc, u16* Uzt, u16* Urt, u16* Uht) {
    int idx = blockIdx.x * 256 + threadIdx.x;        // < 393216
    if (idx < 196608) {
        int n = idx >> 8, k = idx & 255;
        const float* W = (n < 256) ? Wz : (n < 512 ? Wr : Wh);
        int nn = n & 255;
        Wt[idx] = f2bf(W[k * 256 + nn]);
        if (k == 0) {
            const float* bb = (n < 256) ? bz : (n < 512 ? br : bh);
            bc[n] = bb[nn];
        }
    } else {
        int j = idx - 196608;
        int which = j >> 16;                          // 0=Uz 1=Ur 2=Uh
        int n = (j >> 8) & 255, k = j & 255;
        const float* U = (which == 0) ? Uz : (which == 1 ? Ur : Uh);
        u16* Ut = (which == 0) ? Uzt : (which == 1 ? Urt : Uht);
        Ut[n * 256 + k] = f2bf(U[k * 256 + n]);
    }
}

// ---------------- phase 1: P[m][0:768] = x @ [Wz|Wr|Wh] + bias, m = t*64+b -----------
__global__ __launch_bounds__(256) void k_proj(const u16* __restrict__ Xb,
                                              const u16* __restrict__ Wt,
                                              const float* __restrict__ bc,
                                              float* __restrict__ P) {
    int gw = (blockIdx.x * 256 + threadIdx.x) >> 5;   // global wave id, < 2048*12
    int lane = threadIdx.x & 31, hl = lane >> 4;
    int m0 = (gw / 12) * 16, ng = (gw % 12) * 64;
    f32x8 acc[4];
#pragma unroll
    for (int j = 0; j < 4; ++j) acc[j] = (f32x8){0.f,0.f,0.f,0.f,0.f,0.f,0.f,0.f};
#pragma unroll
    for (int kk = 0; kk < 8; ++kk) {
        BF16Vec a = load_a16(Xb, 256, m0, kk);
#pragma unroll
        for (int j = 0; j < 4; ++j) {
            BF16Vec b = load_b16(Wt, 256, ng + j * 16, kk);
            acc[j] = wmma_bf16(a, b, acc[j]);
        }
    }
#pragma unroll
    for (int j = 0; j < 4; ++j)
#pragma unroll
        for (int v = 0; v < 8; ++v) {
            int m = m0 + v + hl * 8, n = ng + j * 16 + (lane & 15);
            P[m * 768 + n] = acc[j][v] + bc[n];
        }
}

// ---------------- phase 2: persistent single-WGP recurrence over T ----------------
__global__ __launch_bounds__(1024) void k_rnn(const int* __restrict__ dep,
                                              const int* __restrict__ mask,
                                              const float* __restrict__ init,
                                              const u16* __restrict__ Urt,
                                              const u16* __restrict__ Uzt,
                                              const u16* __restrict__ Uht,
                                              const float* __restrict__ P,
                                              float* __restrict__ buf,
                                              u16* __restrict__ bufb,
                                              float* __restrict__ out) {
    __shared__ u16 Sb[4 * PLANE];    // 132 KB: 4 state planes, bf16, TDM-padded rows
    __shared__ u16 rsb[PLANE];       //  33 KB: (r*states).sum, bf16, padded rows
    __shared__ u16 UrtL[256 * SP];   // 132 KB: Ur^T staged once via TDM
    const int tid = threadIdx.x;
    const int lane = tid & 31, wv = tid >> 5, hl = lane >> 4;
    const uint32_t sb_lds  = (uint32_t)(uintptr_t)(void*)&Sb[0];
    const uint32_t urt_lds = (uint32_t)(uintptr_t)(void*)&UrtL[0];

    // stage Ur^T into LDS once: one TDM descriptor (256 rows x 256 cols, hw row padding)
    if (wv == 0) {
        tdm_load_2d(urt_lds, Urt, 256);
        __builtin_amdgcn_s_wait_tensorcnt(0);
    }
    __syncthreads();

    for (int t = 0; t < NT; ++t) {
        // ---- stage 4 state planes into LDS + prefetch P for t+2 ----
        if (t + 2 < NT) {
            const char* pp = (const char*)(P + ((size_t)((t + 2) << 6)) * 768);
            __builtin_prefetch(pp + tid * 192, 0, 1);   // 1024 threads x 192B = 192KB slice
        }
        if (t == 0) {
            for (int i = tid; i < BU; i += 1024) {
                int b = i >> 8, u = i & 255, o = b * SP + u;
                Sb[o]             = f2bf(init[i]);
                Sb[PLANE + o]     = f2bf(init[BU + i]);
                Sb[2 * PLANE + o] = f2bf(init[2 * BU + i]);
                Sb[3 * PLANE + o] = f2bf(init[3 * BU + i]);
            }
        } else if (wv == 0) {
            int d0 = dep[(t - 1) * 3], d1 = dep[(t - 1) * 3 + 1], d2 = dep[(t - 1) * 3 + 2];
            tdm_load_2d(sb_lds,                  bufb + (size_t)(t - 1) * BU, 64);
            tdm_load_2d(sb_lds + PLANE * 2,      bufb + (size_t)d0 * BU, 64);
            tdm_load_2d(sb_lds + 2 * PLANE * 2,  bufb + (size_t)d1 * BU, 64);
            tdm_load_2d(sb_lds + 3 * PLANE * 2,  bufb + (size_t)d2 * BU, 64);
            __builtin_amdgcn_s_wait_tensorcnt(0);
        }
        __syncthreads();

        // ---- G1: r_k = sigmoid(Pr + S_k @ Ur); rs = sum_k r_k * S_k (wave-local tiles) ----
        for (int g = wv; g < 64; g += 32) {
            int b0 = (g >> 4) * 16, n0 = (g & 15) * 16;
            float pr[8], rsacc[8];
#pragma unroll
            for (int v = 0; v < 8; ++v) {
                int bb = b0 + v + hl * 8, u = n0 + (lane & 15);
                pr[v] = P[((t << 6) + bb) * 768 + 256 + u];
                rsacc[v] = 0.f;
            }
#pragma unroll
            for (int k = 0; k < 4; ++k) {
                f32x8 acc = {0.f, 0.f, 0.f, 0.f, 0.f, 0.f, 0.f, 0.f};
#pragma unroll
                for (int kk = 0; kk < 8; ++kk) {
                    BF16Vec a = load_a16(Sb + k * PLANE, SP, b0, kk);
                    BF16Vec b = load_b16(UrtL, SP, n0, kk);
                    acc = wmma_bf16(a, b, acc);
                }
#pragma unroll
                for (int v = 0; v < 8; ++v) {
                    int bb = b0 + v + hl * 8, u = n0 + (lane & 15);
                    float r = sigm(acc[v] + pr[v]);
                    rsacc[v] += r * bf2f(Sb[k * PLANE + bb * SP + u]);
                }
            }
#pragma unroll
            for (int v = 0; v < 8; ++v) {
                int bb = b0 + v + hl * 8, u = n0 + (lane & 15);
                rsb[bb * SP + u] = f2bf(rsacc[v]);
            }
        }
        __syncthreads();

        // ---- G2: z = sigmoid(Pz + h_sum@Uz); G3: h~ = tanh(Ph + rs@Uh); combine ----
        for (int g = wv; g < 64; g += 32) {
            int b0 = (g >> 4) * 16, n0 = (g & 15) * 16;
            f32x8 accz = {0.f, 0.f, 0.f, 0.f, 0.f, 0.f, 0.f, 0.f};
            f32x8 acch = {0.f, 0.f, 0.f, 0.f, 0.f, 0.f, 0.f, 0.f};
#pragma unroll
            for (int kk = 0; kk < 8; ++kk) {
                BF16Vec az = load_a_sum4(Sb, b0, kk);
                BF16Vec bz = load_b16(Uzt, 256, n0, kk);
                accz = wmma_bf16(az, bz, accz);
                BF16Vec ar = load_a16(rsb, SP, b0, kk);
                BF16Vec bh = load_b16(Uht, 256, n0, kk);
                acch = wmma_bf16(ar, bh, acch);
            }
#pragma unroll
            for (int v = 0; v < 8; ++v) {
                int bb = b0 + v + hl * 8, u = n0 + (lane & 15);
                float pz = P[((t << 6) + bb) * 768 + u];
                float ph = P[((t << 6) + bb) * 768 + 512 + u];
                float z  = sigm(accz[v] + pz);
                float ht = tanh_(acch[v] + ph);
                float hs = bf2f(Sb[bb * SP + u]) + bf2f(Sb[PLANE + bb * SP + u]) +
                           bf2f(Sb[2 * PLANE + bb * SP + u]) + bf2f(Sb[3 * PLANE + bb * SP + u]);
                float h  = z * hs * 0.25f + (1.f - z) * ht;
                int   mk = mask[bb * NT + t];
                float prev = (t == 0) ? 0.f : buf[(t - 1) * BU + bb * 256 + u];
                float st = mk ? h : prev;
                float o  = mk ? h : 0.f;
                buf[t * BU + bb * 256 + u]  = st;          // f32 ring (exact masked carry)
                bufb[t * BU + bb * 256 + u] = f2bf(st);    // bf16 mirror for TDM staging
                out[(bb * NT + t) * NU + u] = o;
                if (t == NT - 1) {
                    out[OUT_LO + bb * 256 + u] = o;
                    out[OUT_LS + bb * 256 + u] = st;
                }
            }
        }
        __threadfence();   // make buf/bufb visible to next step's TDM + loads
        __syncthreads();
    }
}

extern "C" void kernel_launch(void* const* d_in, const int* in_sizes, int n_in,
                              void* d_out, int out_size, void* d_ws, size_t ws_size,
                              hipStream_t stream) {
    (void)in_sizes; (void)n_in; (void)out_size; (void)ws_size;
    const float* inputs = (const float*)d_in[0];
    const int*   dep    = (const int*)d_in[1];
    const int*   mask   = (const int*)d_in[2];
    const float* init   = (const float*)d_in[3];
    const float* Wz = (const float*)d_in[4];
    const float* Wr = (const float*)d_in[5];
    const float* Wh = (const float*)d_in[6];
    const float* Uz = (const float*)d_in[7];
    const float* Ur = (const float*)d_in[8];
    const float* Uh = (const float*)d_in[9];
    const float* bz = (const float*)d_in[10];
    const float* br = (const float*)d_in[11];
    const float* bh = (const float*)d_in[12];
    float* out = (float*)d_out;

    char* ws = (char*)d_ws;
    size_t off = 0;
    u16* Xb  = (u16*)(ws + off); off += (size_t)32768 * 256 * 2;   // 16 MB
    u16* Wt  = (u16*)(ws + off); off += (size_t)768 * 256 * 2;
    u16* Uzt = (u16*)(ws + off); off += (size_t)256 * 256 * 2;
    u16* Urt = (u16*)(ws + off); off += (size_t)256 * 256 * 2;
    u16* Uht = (u16*)(ws + off); off += (size_t)256 * 256 * 2;
    float* bc = (float*)(ws + off); off += (size_t)768 * 4;
    float* P  = (float*)(ws + off); off += (size_t)32768 * 768 * 4; // ~100 MB
    float* buf  = (float*)(ws + off); off += (size_t)NT * BU * 4;   // 32 MB f32 ring
    u16*   bufb = (u16*)(ws + off);   off += (size_t)NT * BU * 2;   // 16 MB bf16 mirror

    k_cvt_x<<<32768, 256, 0, stream>>>(inputs, Xb);
    k_cvt_w<<<1536, 256, 0, stream>>>(Wz, Wr, Wh, Uz, Ur, Uh, bz, br, bh,
                                      Wt, bc, Uzt, Urt, Uht);
    k_proj<<<3072, 256, 0, stream>>>(Xb, Wt, bc, P);   // 2048 M-tiles x 12 N-groups
    k_rnn<<<1, 1024, 0, stream>>>(dep, mask, init, Urt, Uzt, Uht, P, buf, bufb, out);
}
